// QKVAttention_50388556317336
// MI455X (gfx1250) — matrix operand order repaired and checked by reading, not compiled, not run on hardware
//
#include <hip/hip_runtime.h>
#include <stdint.h>

// ---- trivially-copyable vector types (safe inside unions) ----
typedef __attribute__((ext_vector_type(16))) __bf16   v16bf;
typedef __attribute__((ext_vector_type(8)))  float    v8f;
typedef __attribute__((ext_vector_type(4)))  uint32_t u32x4;
typedef __attribute__((ext_vector_type(4)))  __bf16   bf16x4;
typedef __attribute__((ext_vector_type(4)))  float    f32x4;

union Frag {
  v16bf    v;
  u32x4    q[2];
  uint32_t u[8];
};

#define WMMA_BF16(A, B, Cacc) \
  __builtin_amdgcn_wmma_f32_16x16x32_bf16(false, (A), false, (B), (short)0, (Cacc), false, false)

#define EXP2F(x) __builtin_amdgcn_exp2f(x)   // v_exp_f32 (hardware exp2)

// ---- 16-lane butterfly reductions in pure VALU via DPP16 ----
// xor1 = quad_perm[1,0,3,2] (0xB1), xor2 = quad_perm[2,3,0,1] (0x4E),
// xor7 = row_half_mirror (0x141),   xor15 = row_mirror (0x140).
// Combining all four fully reduces each 16-lane half (rows M=r / M=r+8 stay separate).
#define DPP_STEP_MAX(x, ctrl)                                                   \
  x = fmaxf(x, __int_as_float(__builtin_amdgcn_mov_dpp(                         \
                  __float_as_int(x), (ctrl), 0xF, 0xF, true)))
#define DPP_STEP_ADD(x, ctrl)                                                   \
  x = x + __int_as_float(__builtin_amdgcn_mov_dpp(                              \
                  __float_as_int(x), (ctrl), 0xF, 0xF, true))

__device__ __forceinline__ float reduce_max16(float x) {
  DPP_STEP_MAX(x, 0xB1);   // quad_perm [1,0,3,2]
  DPP_STEP_MAX(x, 0x4E);   // quad_perm [2,3,0,1]
  DPP_STEP_MAX(x, 0x141);  // row_half_mirror
  DPP_STEP_MAX(x, 0x140);  // row_mirror
  return x;
}
__device__ __forceinline__ float reduce_sum16(float x) {
  DPP_STEP_ADD(x, 0xB1);
  DPP_STEP_ADD(x, 0x4E);
  DPP_STEP_ADD(x, 0x141);
  DPP_STEP_ADD(x, 0x140);
  return x;
}

// qkv: [4, 1536, 2048] fp32; out: [4, 512, 2048] fp32
// Per head: Q,K,V are [64 c][2048 t] (t contiguous). Flash attention, bf16 WMMA,
// softmax computed in the exp2 domain (log2e folded into the Q scale).
__global__ __launch_bounds__(256)
void qkv_attn_fa_kernel(const float* __restrict__ qkv, float* __restrict__ out) {
  constexpr int L  = 2048;
  constexpr int C  = 64;
  constexpr int W3 = 1536;
  constexpr int HC = 512;
  constexpr float QSCALE = 0.125f * 1.4426950408889634f;  // 1/sqrt(64) * log2(e)

  const int tileT = blockIdx.x & 15;   // 16 T-tiles of 128
  const int bh    = blockIdx.x >> 4;   // 0..31  (bs*heads)
  const int bsi   = bh >> 3;
  const int hd    = bh & 7;

  const float* qp = qkv + (size_t)bsi * W3 * L + (size_t)(hd * C) * L;
  const float* kp = qp + (size_t)HC * L;
  const float* vp = qp + (size_t)(2 * HC) * L;
  float*       op = out + (size_t)bsi * HC * L + (size_t)(hd * C) * L;

  const int tid  = threadIdx.x;
  const int wave = tid >> 5;
  const int lane = tid & 31;
  const int half = lane >> 4;
  const int col  = lane & 15;
  const int t0   = tileT * 128;

  __shared__ __align__(16) __bf16 Qt[128][64];    // Q^T tile [t][c], pre-scaled
  __shared__ __align__(16) __bf16 Kt[64][64];     // K^T tile [s][c]
  __shared__ __align__(16) __bf16 Vt[64][64];     // V tile   [c][s]
  __shared__ __align__(16) __bf16 Pb[8][16][64];  // per-wave P round-trip [t][s]

  // ---- stage Q^T, scaled ----
  for (int i = tid; i < 2048; i += 256) {           // 2048 float4 chunks
    int c  = i >> 5;
    int t4 = (i & 31) << 2;
    f32x4 f = *(const f32x4*)&qp[(size_t)c * L + t0 + t4];
    Qt[t4 + 0][c] = (__bf16)(f.x * QSCALE);
    Qt[t4 + 1][c] = (__bf16)(f.y * QSCALE);
    Qt[t4 + 2][c] = (__bf16)(f.z * QSCALE);
    Qt[t4 + 3][c] = (__bf16)(f.w * QSCALE);
  }
  __syncthreads();

  // ---- Q A-fragments (held for the whole S loop) ----
  Frag a0, a1;
  {
    const int trow = wave * 16 + col;
    const int klo  = half * 8;
    a0.q[0] = *(const u32x4*)&Qt[trow][klo];
    a0.q[1] = *(const u32x4*)&Qt[trow][klo + 16];
    a1.q[0] = *(const u32x4*)&Qt[trow][32 + klo];
    a1.q[1] = *(const u32x4*)&Qt[trow][32 + klo + 16];
  }

  // ---- state: 8 rows per lane (M = r + 8*half). lsum kept UNREDUCED per lane.
  float m[8], lsum[8];
  v8f   o[4];
  v8f   zero8 = {};
#pragma unroll
  for (int r = 0; r < 8; ++r) { m[r] = -1e30f; lsum[r] = 0.f; }
#pragma unroll
  for (int ct = 0; ct < 4; ++ct) o[ct] = zero8;

  for (int s0 = 0; s0 < L; s0 += 64) {
    __syncthreads();
    // ---- stage K^T [s][c] and V [c][s] in bf16 ----
    for (int i = tid; i < 1024; i += 256) {
      int c  = i >> 4;
      int s4 = (i & 15) << 2;
      f32x4 kf = *(const f32x4*)&kp[(size_t)c * L + s0 + s4];
      f32x4 vf = *(const f32x4*)&vp[(size_t)c * L + s0 + s4];
      Kt[s4 + 0][c] = (__bf16)kf.x;
      Kt[s4 + 1][c] = (__bf16)kf.y;
      Kt[s4 + 2][c] = (__bf16)kf.z;
      Kt[s4 + 3][c] = (__bf16)kf.w;
      bf16x4 pv = { (__bf16)vf.x, (__bf16)vf.y, (__bf16)vf.z, (__bf16)vf.w };
      *(bf16x4*)&Vt[c][s4] = pv;
    }
    __syncthreads();

    // prefetch next S-tile of K/V (global_prefetch_b8)
    if (s0 + 64 < L) {
      int pi = wave * 32 + lane;
      int c  = (pi >> 2) & 63;
      int ln = (pi & 1) * 32;
      const float* base = (pi & 2) ? vp : kp;
      __builtin_prefetch(&base[(size_t)c * L + s0 + 64 + ln], 0, 0);
    }

    // ---- scores: S[16t x 64s] = Q^T * K ----
    v8f sc[4];
#pragma unroll
    for (int j = 0; j < 4; ++j) {
      Frag b0, b1;
      const int srow = j * 16 + col;
      const int cb   = half * 8;
      b0.q[0] = *(const u32x4*)&Kt[srow][cb];
      b0.q[1] = *(const u32x4*)&Kt[srow][cb + 16];
      b1.q[0] = *(const u32x4*)&Kt[srow][32 + cb];
      b1.q[1] = *(const u32x4*)&Kt[srow][32 + cb + 16];
      v8f acc = zero8;
      acc = WMMA_BF16(a0.v, b0.v, acc);
      acc = WMMA_BF16(a1.v, b1.v, acc);
      sc[j] = acc;
    }

    // ---- online max (exp2 domain); sums stay per-lane (reduced once at end) ----
    float alpha[8];
#pragma unroll
    for (int r = 0; r < 8; ++r) {
      float t = fmaxf(fmaxf(sc[0][r], sc[1][r]), fmaxf(sc[2][r], sc[3][r]));
      t = reduce_max16(t);
      float nm = fmaxf(m[r], t);
      alpha[r] = EXP2F(m[r] - nm);
      m[r] = nm;
    }
#pragma unroll
    for (int r = 0; r < 8; ++r) {
      float rs = 0.f;
#pragma unroll
      for (int j = 0; j < 4; ++j) {
        float p = EXP2F(sc[j][r] - m[r]);
        rs += p;
        Pb[wave][r + 8 * half][j * 16 + col] = (__bf16)p;
      }
      lsum[r] = lsum[r] * alpha[r] + rs;   // per-lane partial of this row's sum
    }
#pragma unroll
    for (int ct = 0; ct < 4; ++ct)
#pragma unroll
      for (int r = 0; r < 8; ++r) o[ct][r] *= alpha[r];

    // ---- reload P as A-fragments (C/D -> A transpose via per-wave LDS) ----
    Frag pa0, pa1;
    {
      const int klo = half * 8;
      pa0.q[0] = *(const u32x4*)&Pb[wave][col][klo];
      pa0.q[1] = *(const u32x4*)&Pb[wave][col][klo + 16];
      pa1.q[0] = *(const u32x4*)&Pb[wave][col][32 + klo];
      pa1.q[1] = *(const u32x4*)&Pb[wave][col][32 + klo + 16];
    }

    // ---- O[16t x 64c] += P[16t x 64s] * V^T[64s x 64c] ----
#pragma unroll
    for (int ct = 0; ct < 4; ++ct) {
      Frag bv0, bv1;
      const int crow = ct * 16 + col;
      const int sb   = half * 8;
      bv0.q[0] = *(const u32x4*)&Vt[crow][sb];
      bv0.q[1] = *(const u32x4*)&Vt[crow][sb + 16];
      bv1.q[0] = *(const u32x4*)&Vt[crow][32 + sb];
      bv1.q[1] = *(const u32x4*)&Vt[crow][32 + sb + 16];
      o[ct] = WMMA_BF16(pa0.v, bv0.v, o[ct]);
      o[ct] = WMMA_BF16(pa1.v, bv1.v, o[ct]);
    }
  }

  // ---- single deferred reduction of the row sums, then normalize + store ----
  float inv[8];
#pragma unroll
  for (int r = 0; r < 8; ++r) inv[r] = 1.0f / reduce_sum16(lsum[r]);
#pragma unroll
  for (int ct = 0; ct < 4; ++ct) {
    const int c = ct * 16 + col;
#pragma unroll
    for (int r = 0; r < 8; ++r) {
      const int t = t0 + wave * 16 + r + 8 * half;
      op[(size_t)c * L + t] = o[ct][r] * inv[r];
    }
  }
}

extern "C" void kernel_launch(void* const* d_in, const int* in_sizes, int n_in,
                              void* d_out, int out_size, void* d_ws, size_t ws_size,
                              hipStream_t stream) {
  const float* qkv = (const float*)d_in[0];
  // d_in[1] = num_heads (==8); dims fixed by harness setup: [4, 1536, 2048]
  float* out = (float*)d_out;
  dim3 grid(32 * 16);   // 32 batch-heads x 16 T-tiles of 128 rows
  dim3 block(256);      // 8 waves (wave32)
  hipLaunchKernelGGL(qkv_attn_fa_kernel, grid, block, 0, stream, qkv, out);
}
